// RNNModel_18116172054930
// MI455X (gfx1250) — compile-verified, matrix operand order checked by reading, not run on hardware
//
#include <hip/hip_runtime.h>
#include <stddef.h>
#include <stdint.h>

// ---------------------------------------------------------------------------
// Problem constants: B=64, T=512, D_IN=256, H=512, D_OUT=128
// ---------------------------------------------------------------------------
#define BB      64
#define TT      512
#define D_IN    256
#define HH      512
#define D_OUT   128
#define MTOT    (BB * TT)          // 32768 rows for all big GEMMs

// ---------------------------------------------------------------------------
// CDNA5 WMMA types
// ---------------------------------------------------------------------------
typedef __attribute__((ext_vector_type(16))) __bf16 v16bf;
typedef __attribute__((ext_vector_type(8)))  __bf16 v8bf;
typedef __attribute__((ext_vector_type(8)))  float  v8f;
typedef __attribute__((ext_vector_type(4)))  int    v4i;

#define AS_GLOBAL __attribute__((address_space(1)))
#define AS_LDS    __attribute__((address_space(3)))

union BF16Frag {
    v16bf v;
    v8bf  h[2];
};

static __device__ __forceinline__ v8f vzero8() {
    v8f z = {0.f, 0.f, 0.f, 0.f, 0.f, 0.f, 0.f, 0.f};
    return z;
}

static __device__ __forceinline__ unsigned short f32_to_bf16(float f) {
    unsigned int u = __float_as_uint(f);
    unsigned int r = u + 0x7FFFu + ((u >> 16) & 1u);   // round-to-nearest-even
    return (unsigned short)(r >> 16);
}

// ---------------------------------------------------------------------------
// gfx1250 async global->LDS copy (ASYNCcnt path), with sync fallback
// ---------------------------------------------------------------------------
#if defined(__gfx1250__) && __has_builtin(__builtin_amdgcn_global_load_async_to_lds_b128)
#define HAVE_ASYNC_LDS 1
#else
#define HAVE_ASYNC_LDS 0
#endif

static __device__ __forceinline__ void async_copy16(const unsigned short* g,
                                                    unsigned short* l) {
#if HAVE_ASYNC_LDS
    __builtin_amdgcn_global_load_async_to_lds_b128(
        (AS_GLOBAL v4i*)g,      // global source (param type: int4 __device__*)
        (AS_LDS v4i*)l,         // LDS destination
        0, 0);                  // imm offset, cpol
#else
    *(v8bf*)l = *(const v8bf*)g;
#endif
}

static __device__ __forceinline__ void async_wait0() {
#if HAVE_ASYNC_LDS && __has_builtin(__builtin_amdgcn_s_wait_asynccnt)
    __builtin_amdgcn_s_wait_asynccnt(0);
#endif
}

// ---------------------------------------------------------------------------
// f32 -> bf16 conversion (grid-stride)
// ---------------------------------------------------------------------------
__global__ void cvt_bf16(const float* __restrict__ in,
                         unsigned short* __restrict__ out, size_t n) {
    size_t stride = (size_t)gridDim.x * blockDim.x;
    for (size_t i = (size_t)blockIdx.x * blockDim.x + threadIdx.x; i < n; i += stride)
        out[i] = f32_to_bf16(in[i]);
}

// ---------------------------------------------------------------------------
// Tiled bf16 WMMA GEMM: C[M,N] = A[M,K] @ W[N,K]^T + b0[N] (+ b1[N])
// Block tile 128(M) x 128(N), K-step 32, 256 threads = 8 waves.
// Wave = 32(M) x 64(N) sub-tile -> 2x4 WMMA tiles, 8 accumulators.
// Double-buffered LDS fed by async global->LDS loads.
// Requires M%128==0, N%128==0, K%32==0 (true for all uses here).
// ---------------------------------------------------------------------------
#define GA_STRIDE 40   // 32 + 8 pad: 16B-aligned rows, conflict-free b128 reads

__global__ __launch_bounds__(256)
void wmma_gemm(const unsigned short* __restrict__ A,
               const unsigned short* __restrict__ W,
               const float* __restrict__ b0,
               const float* __restrict__ b1,
               float* __restrict__ C,
               int M, int N, int K) {
    __shared__ __align__(16) unsigned short lA[2][128 * GA_STRIDE];
    __shared__ __align__(16) unsigned short lB[2][128 * GA_STRIDE];

    const int tid  = threadIdx.x;
    const int lane = tid & 31;
    const int wave = tid >> 5;
    const int wm   = wave >> 1;      // 0..3 -> 32-row strip
    const int wn   = wave & 1;       // 0..1 -> 64-col strip

    const int mBase = blockIdx.y * 128;
    const int nBase = blockIdx.x * 128;

    const int fRow = lane & 15;          // fragment row (M for A, N for B^T)
    const int kb   = (lane >> 4) * 8;    // per-lane-half K sub-offset

    // Staging chunk coordinates: 128 rows x 32 cols = 512 x 16B chunks,
    // 256 threads -> 2 chunks each per tile.
    const int r0 = tid >> 2,         o0 = (tid & 3) * 8;
    const int r1 = (tid + 256) >> 2, o1 = ((tid + 256) & 3) * 8;

    auto stage = [&](int buf, int kt) {
        async_copy16(&A[(size_t)(mBase + r0) * K + kt + o0], &lA[buf][r0 * GA_STRIDE + o0]);
        async_copy16(&A[(size_t)(mBase + r1) * K + kt + o1], &lA[buf][r1 * GA_STRIDE + o1]);
        async_copy16(&W[(size_t)(nBase + r0) * K + kt + o0], &lB[buf][r0 * GA_STRIDE + o0]);
        async_copy16(&W[(size_t)(nBase + r1) * K + kt + o1], &lB[buf][r1 * GA_STRIDE + o1]);
    };

    v8f acc[2][4];
#pragma unroll
    for (int mt = 0; mt < 2; ++mt)
#pragma unroll
        for (int nt = 0; nt < 4; ++nt)
            acc[mt][nt] = vzero8();

    stage(0, 0);                       // prologue: fill buffer 0
    int buf = 0;
    for (int kt = 0; kt < K; kt += 32) {
        async_wait0();                 // my async loads into lA/lB[buf] done
        __syncthreads();               // everyone's loads done, prev compute done
        if (kt + 32 < K) stage(buf ^ 1, kt + 32);   // prefetch next tile

        BF16Frag afr[2], bfr[4];
#pragma unroll
        for (int mt = 0; mt < 2; ++mt) {
            const unsigned short* p = &lA[buf][(wm * 32 + mt * 16 + fRow) * GA_STRIDE + kb];
            afr[mt].h[0] = *(const v8bf*)(p);
            afr[mt].h[1] = *(const v8bf*)(p + 16);
        }
#pragma unroll
        for (int nt = 0; nt < 4; ++nt) {
            const unsigned short* p = &lB[buf][(wn * 64 + nt * 16 + fRow) * GA_STRIDE + kb];
            bfr[nt].h[0] = *(const v8bf*)(p);
            bfr[nt].h[1] = *(const v8bf*)(p + 16);
        }
#pragma unroll
        for (int mt = 0; mt < 2; ++mt)
#pragma unroll
            for (int nt = 0; nt < 4; ++nt)
                acc[mt][nt] = __builtin_amdgcn_wmma_f32_16x16x32_bf16(
                    false, afr[mt].v, false, bfr[nt].v,
                    (short)0, acc[mt][nt], false, false);
        buf ^= 1;
    }

    // Epilogue: bias + f32 store. D layout: lane = (m>=8)*16 + n, vgpr r = m&7.
#pragma unroll
    for (int mt = 0; mt < 2; ++mt) {
#pragma unroll
        for (int nt = 0; nt < 4; ++nt) {
            int n = nBase + wn * 64 + nt * 16 + (lane & 15);
            float bias = b0[n] + (b1 ? b1[n] : 0.0f);
#pragma unroll
            for (int r = 0; r < 8; ++r) {
                int m = mBase + wm * 32 + mt * 16 + (lane >> 4) * 8 + r;
                C[(size_t)m * N + n] = acc[mt][nt][r] + bias;
            }
        }
    }
}

// ---------------------------------------------------------------------------
// Recurrent scan:  h_t = tanh( xW[:,t,:] + h_{t-1} @ W_hh^T )
// One block per 16-batch tile (grid = 4), 512 threads = 16 waves; wave w owns
// output columns [32w, 32w+32). h_prev (16 x 512 bf16) lives in LDS.
// W_hh for K=0..255 is held permanently in VGPRs (t-invariant, 128 VGPRs/lane)
// so each step's first 16 WMMAs start right after the barrier with zero global
// latency; K=256..511 streams from L2 underneath them.
// ---------------------------------------------------------------------------
#define HP_STRIDE 520   // 512 + 8 pad, keeps 16B alignment, conflict-free

__global__ __launch_bounds__(512)
void rnn_scan(const float* __restrict__ xW,           // [B*T, 512] f32
              const unsigned short* __restrict__ Whh, // [512, 512] bf16 (B^T)
              unsigned short* __restrict__ hOut) {    // [B*T, 512] bf16
    __shared__ __align__(16) unsigned short hPrev[16 * HP_STRIDE];

    const int tid  = threadIdx.x;
    const int lane = tid & 31;
    const int wave = tid >> 5;        // 0..15
    const int bb   = blockIdx.x;      // 0..3 batch tile

    for (int i = tid; i < 16 * HP_STRIDE; i += 512) hPrev[i] = 0;
    __syncthreads();

    const int fRow = lane & 15;
    const int kb   = (lane >> 4) * 8;
    const int nLoc = wave * 32;                    // wave's first local column
    const int n0   = nLoc + fRow;                  // nt=0 fragment column
    const int n1   = nLoc + 16 + fRow;             // nt=1 fragment column
    const unsigned short* wrow0 = &Whh[(size_t)n0 * HH];
    const unsigned short* wrow1 = &Whh[(size_t)n1 * HH];

    // Register-resident W fragments for K = 0..255 (8 k-tiles x 2 col-tiles)
    BF16Frag wlo[2][8];
#pragma unroll
    for (int k8 = 0; k8 < 8; ++k8) {
        wlo[0][k8].h[0] = *(const v8bf*)&wrow0[k8 * 32 + kb];
        wlo[0][k8].h[1] = *(const v8bf*)&wrow0[k8 * 32 + kb + 16];
        wlo[1][k8].h[0] = *(const v8bf*)&wrow1[k8 * 32 + kb];
        wlo[1][k8].h[1] = *(const v8bf*)&wrow1[k8 * 32 + kb + 16];
    }

    const int  mloc0 = (lane >> 4) * 8;
    const size_t rowStep = (size_t)TT * HH;        // xW/hOut row stride per batch
    const float* xwB = &xW[(size_t)(bb * 16) * rowStep];
    unsigned short* hoB = &hOut[(size_t)(bb * 16) * rowStep];

    for (int t = 0; t < TT; ++t) {
        v8f acc[2];
        // C init = input projection for this (batch-tile, t)
#pragma unroll
        for (int r = 0; r < 8; ++r) {
            const float* p = &xwB[(size_t)(mloc0 + r) * rowStep + (size_t)t * HH];
            acc[0][r] = p[n0];
            acc[1][r] = p[n1];
        }
        // K loop: A-fragment shared by both column tiles (one LDS load per kt)
#pragma unroll
        for (int kt = 0; kt < 16; ++kt) {
            BF16Frag a;
            const unsigned short* hp = &hPrev[fRow * HP_STRIDE + kt * 32 + kb];
            a.h[0] = *(const v8bf*)(hp);
            a.h[1] = *(const v8bf*)(hp + 16);
            if (kt < 8) {
                acc[0] = __builtin_amdgcn_wmma_f32_16x16x32_bf16(
                    false, a.v, false, wlo[0][kt].v, (short)0, acc[0], false, false);
                acc[1] = __builtin_amdgcn_wmma_f32_16x16x32_bf16(
                    false, a.v, false, wlo[1][kt].v, (short)0, acc[1], false, false);
            } else {
                BF16Frag w0, w1;
                w0.h[0] = *(const v8bf*)&wrow0[kt * 32 + kb];
                w0.h[1] = *(const v8bf*)&wrow0[kt * 32 + kb + 16];
                w1.h[0] = *(const v8bf*)&wrow1[kt * 32 + kb];
                w1.h[1] = *(const v8bf*)&wrow1[kt * 32 + kb + 16];
                acc[0] = __builtin_amdgcn_wmma_f32_16x16x32_bf16(
                    false, a.v, false, w0.v, (short)0, acc[0], false, false);
                acc[1] = __builtin_amdgcn_wmma_f32_16x16x32_bf16(
                    false, a.v, false, w1.v, (short)0, acc[1], false, false);
            }
        }

        float hNew[2][8];
#pragma unroll
        for (int nt = 0; nt < 2; ++nt)
#pragma unroll
            for (int r = 0; r < 8; ++r) hNew[nt][r] = tanhf(acc[nt][r]);

        __syncthreads();   // everyone done reading h_{t-1}
#pragma unroll
        for (int r = 0; r < 8; ++r) {
            int mloc = mloc0 + r;
            unsigned short hb0 = f32_to_bf16(hNew[0][r]);
            unsigned short hb1 = f32_to_bf16(hNew[1][r]);
            hPrev[mloc * HP_STRIDE + n0] = hb0;
            hPrev[mloc * HP_STRIDE + n1] = hb1;
            unsigned short* q = &hoB[(size_t)mloc * rowStep + (size_t)t * HH];
            q[n0] = hb0;
            q[n1] = hb1;
        }
        __syncthreads();   // h_t visible before next step's reads
    }
}

// ---------------------------------------------------------------------------
// Launcher
// ---------------------------------------------------------------------------
extern "C" void kernel_launch(void* const* d_in, const int* in_sizes, int n_in,
                              void* d_out, int out_size, void* d_ws, size_t ws_size,
                              hipStream_t stream) {
    (void)in_sizes; (void)n_in; (void)out_size; (void)ws_size;

    const float* x     = (const float*)d_in[0];
    const float* W_ih0 = (const float*)d_in[1];
    const float* W_hh0 = (const float*)d_in[2];
    const float* b_ih0 = (const float*)d_in[3];
    const float* b_hh0 = (const float*)d_in[4];
    const float* W_ih1 = (const float*)d_in[5];
    const float* W_hh1 = (const float*)d_in[6];
    const float* b_ih1 = (const float*)d_in[7];
    const float* b_hh1 = (const float*)d_in[8];
    const float* W_fc  = (const float*)d_in[9];
    const float* b_fc  = (const float*)d_in[10];
    float* out = (float*)d_out;

    // Workspace layout (all offsets 16B-aligned)
    uint8_t* ws = (uint8_t*)d_ws;
    size_t off = 0;
    unsigned short* xb    = (unsigned short*)(ws + off); off += (size_t)MTOT * D_IN * 2;   // 16 MB
    unsigned short* wih0b = (unsigned short*)(ws + off); off += (size_t)HH * D_IN * 2;     // 256 KB
    unsigned short* whh0b = (unsigned short*)(ws + off); off += (size_t)HH * HH * 2;       // 512 KB
    unsigned short* wih1b = (unsigned short*)(ws + off); off += (size_t)HH * HH * 2;       // 512 KB
    unsigned short* whh1b = (unsigned short*)(ws + off); off += (size_t)HH * HH * 2;       // 512 KB
    unsigned short* wfcb  = (unsigned short*)(ws + off); off += (size_t)D_OUT * HH * 2;    // 128 KB
    float*          buf   = (float*)(ws + off);          off += (size_t)MTOT * HH * 4;     // 64 MB (xW0, reused for xW1)
    unsigned short* h1b   = (unsigned short*)(ws + off); off += (size_t)MTOT * HH * 2;     // 32 MB
    unsigned short* h2b   = (unsigned short*)(ws + off); off += (size_t)MTOT * HH * 2;     // 32 MB

    // 1) f32 -> bf16 conversions
    cvt_bf16<<<1024, 256, 0, stream>>>(x,     xb,    (size_t)MTOT * D_IN);
    cvt_bf16<<<256,  256, 0, stream>>>(W_ih0, wih0b, (size_t)HH * D_IN);
    cvt_bf16<<<256,  256, 0, stream>>>(W_hh0, whh0b, (size_t)HH * HH);
    cvt_bf16<<<256,  256, 0, stream>>>(W_ih1, wih1b, (size_t)HH * HH);
    cvt_bf16<<<256,  256, 0, stream>>>(W_hh1, whh1b, (size_t)HH * HH);
    cvt_bf16<<<256,  256, 0, stream>>>(W_fc,  wfcb,  (size_t)D_OUT * HH);

    // 2) Layer-0 input projection: buf = x @ W_ih0^T + b_ih0 + b_hh0
    wmma_gemm<<<dim3(HH / 128, MTOT / 128), 256, 0, stream>>>(
        xb, wih0b, b_ih0, b_hh0, buf, MTOT, HH, D_IN);

    // 3) Layer-0 recurrence -> h1 (bf16)
    rnn_scan<<<BB / 16, 512, 0, stream>>>(buf, whh0b, h1b);

    // 4) Layer-1 input projection: buf = h1 @ W_ih1^T + b_ih1 + b_hh1
    wmma_gemm<<<dim3(HH / 128, MTOT / 128), 256, 0, stream>>>(
        h1b, wih1b, b_ih1, b_hh1, buf, MTOT, HH, HH);

    // 5) Layer-1 recurrence -> h2 (bf16)
    rnn_scan<<<BB / 16, 512, 0, stream>>>(buf, whh1b, h2b);

    // 6) Head: out = h2 @ W_fc^T + b_fc   (f32 to d_out)
    wmma_gemm<<<dim3(D_OUT / 128, MTOT / 128), 256, 0, stream>>>(
        h2b, wfcb, b_fc, nullptr, out, MTOT, D_OUT, HH);
}
